// Graph_Attention_Covlayer_45114336477373
// MI455X (gfx1250) — matrix operand order repaired and checked by reading, not compiled, not run on hardware
//
#include <hip/hip_runtime.h>

#define N_NODES 4096
#define IN_C    256
#define OUT_C   64
#define ALPHA   0.2f

#define KSTEP   128                 // j-columns per producer/consumer step
#define NIT     (N_NODES / KSTEP)   // 32 iterations
#define PROW    136                 // padded pTile row (halfs): 272B = 68 dwords
                                    // -> lane m hits bank 4m (conflict-free b128)

typedef __attribute__((ext_vector_type(16))) _Float16 v16h;
typedef __attribute__((ext_vector_type(8)))  _Float16 v8h;
typedef __attribute__((ext_vector_type(8)))  float    v8f;

// -------------------------------------------------------------------------
// Kernel 1: Wh = H @ W (f32 accumulate), emit WhT in f16 (transposed,
// [OUT_C][N] so B fragments are contiguous-K loads), and f1 = Wh@a1,
// f2 = Wh@a2 via block reduction. One block per graph node, 64 threads.
// -------------------------------------------------------------------------
__global__ __launch_bounds__(64)
void gat_k1(const float* __restrict__ H, const float* __restrict__ W,
            const float* __restrict__ a,
            _Float16* __restrict__ WhT,
            float* __restrict__ f1, float* __restrict__ f2) {
    __shared__ float Hrow[IN_C];
    __shared__ float red[OUT_C];
    const int i = blockIdx.x;
    const int n = threadIdx.x;

    ((float4*)Hrow)[n] = ((const float4*)(H + (size_t)i * IN_C))[n];
    __syncthreads();

    float s = 0.0f;
#pragma unroll 8
    for (int k = 0; k < IN_C; ++k)
        s = fmaf(Hrow[k], W[k * OUT_C + n], s);

    WhT[(size_t)n * N_NODES + i] = (_Float16)s;

    red[n] = s * a[n];
    __syncthreads();
    for (int off = 32; off > 0; off >>= 1) {
        if (n < off) red[n] += red[n + off];
        __syncthreads();
    }
    if (n == 0) f1[i] = red[0];
    __syncthreads();

    red[n] = s * a[OUT_C + n];
    __syncthreads();
    for (int off = 32; off > 0; off >>= 1) {
        if (n < off) red[n] += red[n + off];
        __syncthreads();
    }
    if (n == 0) f2[i] = red[0];
}

// -------------------------------------------------------------------------
// Kernel 2: one block = 16 output rows, 128 threads (4 waves).
// Per K=128 step: every thread computes 16 masked exp-attention values
// (4x int4 adj loads, coalesced) into a double-buffered, bank-padded LDS
// f16 P-tile and privately accumulates the unmasked softmax denominator;
// then each wave (owning 16 output channels) issues 4x
// v_wmma_f32_16x16x32_f16 against L2-resident WhT. One barrier per step.
// -------------------------------------------------------------------------
__global__ __launch_bounds__(128)
void gat_k2(const int* __restrict__ adj,
            const float* __restrict__ f1g, const float* __restrict__ f2g,
            const _Float16* __restrict__ WhT,
            float* __restrict__ out) {
    __shared__ float    f2s[N_NODES];          // 16 KB: full f2 vector
    __shared__ _Float16 pTile[2][16][PROW];    // ~8.5 KB padded P tile
    __shared__ float    f1s[16];
    __shared__ float    Ms[16];                // per-row softmax max
    __shared__ float    red[128];
    __shared__ float    denomP[16][8];
    __shared__ float    dinv[16];

    const int t    = threadIdx.x;
    const int row0 = blockIdx.x * 16;

    // ---- load f2 into LDS (coalesced float4) and find global max ----
    float lmax = -3.4e38f;
    {
        const float4* s4 = (const float4*)f2g;
        float4*       d4 = (float4*)f2s;
        for (int idx = t; idx < N_NODES / 4; idx += 128) {
            float4 v = s4[idx];
            d4[idx]  = v;
            lmax = fmaxf(lmax, fmaxf(fmaxf(v.x, v.y), fmaxf(v.z, v.w)));
        }
    }
    red[t] = lmax;
    __syncthreads();
    for (int off = 64; off > 0; off >>= 1) {
        if (t < off) red[t] = fmaxf(red[t], red[t + off]);
        __syncthreads();
    }
    const float f2max = red[0];
    if (t < 16) {
        float v = f1g[row0 + t];
        f1s[t]  = v;
        float e = v + f2max;                  // lrelu monotone =>
        Ms[t]   = e > 0.0f ? e : ALPHA * e;   // row max = lrelu(f1 + max f2)
    }
    __syncthreads();

    // ---- producer indices: thread t -> (row = t/8, 16 consecutive j) ----
    const int   prow   = t >> 3;
    const int   pk     = (t & 7) * 16;
    const float f1r    = f1s[prow];
    const float Mr     = Ms[prow];
    const int*  adjRow = adj + (size_t)(row0 + prow) * N_NODES + pk;

    // ---- consumer indices (WMMA fragments, wave32) ----
    const int lane = t & 31;
    const int wave = t >> 5;
    const int cn   = lane & 15;   // A: M index, B/C: N index
    const int hi   = lane >> 4;
    const int n0   = wave * 16;   // output-channel tile of this wave
    const _Float16* bPtr = WhT + (size_t)(n0 + cn) * N_NODES + hi * 16;

    v8f   acc        = {};
    float denomLocal = 0.0f;

    for (int it = 0; it < NIT; ++it) {
        const int jb  = it * KSTEP;
        const int buf = it & 1;

        // ---- producer: 16 attention values per thread ----
        const int4*   av  = (const int4*)(adjRow + jb);
        const float4* fv  = (const float4*)(f2s + jb + pk);
        int   mk[16];
        float fl[16];
#pragma unroll
        for (int q = 0; q < 4; ++q) {
            int4   m4 = av[q];
            float4 f4 = fv[q];
            mk[q * 4 + 0] = m4.x; mk[q * 4 + 1] = m4.y;
            mk[q * 4 + 2] = m4.z; mk[q * 4 + 3] = m4.w;
            fl[q * 4 + 0] = f4.x; fl[q * 4 + 1] = f4.y;
            fl[q * 4 + 2] = f4.z; fl[q * 4 + 3] = f4.w;
        }
        if (it + 1 < NIT)                       // uniform branch
            __builtin_prefetch(adjRow + jb + KSTEP, 0, 1);

        v8h h0, h1;
#pragma unroll
        for (int c = 0; c < 16; ++c) {
            float e  = f1r + fl[c];
            e        = e > 0.0f ? e : ALPHA * e;   // LeakyReLU
            float ev = __expf(e - Mr);
            denomLocal += ev;                      // unmasked denominator
            _Float16 hval = (mk[c] > 0) ? (_Float16)ev : (_Float16)0.0f;
            if (c < 8) h0[c] = hval; else h1[c - 8] = hval;
        }
        *(v8h*)&pTile[buf][prow][pk]     = h0;     // 2x ds_store_b128
        *(v8h*)&pTile[buf][prow][pk + 8] = h1;

        __syncthreads();   // P tile visible; other buffer free to rewrite

        // ---- consumer: 4 WMMAs (K = 128) ----
#pragma unroll
        for (int kk = 0; kk < 4; ++kk) {
            // A (contiguous k order chosen here, not the ISA table order): lanes 0-15 K=0..15; lanes 16-31 K=16..31 (matches B)
            v8h alo = *(const v8h*)&pTile[buf][cn][kk * 32 + hi * 16];
            v8h ahi = *(const v8h*)&pTile[buf][cn][kk * 32 + hi * 16 + 8];
            v16h A  = __builtin_shufflevector(alo, ahi, 0, 1, 2, 3, 4, 5, 6, 7,
                                              8, 9, 10, 11, 12, 13, 14, 15);
            // B: lane holds 16 contiguous K values of its output channel
            v16h B = *(const v16h*)(bPtr + jb + kk * 32);

            acc = __builtin_amdgcn_wmma_f32_16x16x32_f16(
                false, A, false, B, (short)0, acc, false, false);
        }
    }

    // ---- reduce denominators (8 partials per row) ----
    denomP[prow][t & 7] = denomLocal;
    __syncthreads();
    if (t < 16) {
        float s = 0.0f;
#pragma unroll
        for (int c = 0; c < 8; ++c) s += denomP[t][c];
        dinv[t] = 1.0f / s;
    }
    __syncthreads();

    // ---- epilogue: C layout = lane->N, VGPR r -> M=r (lanes<16) / 8+r ----
#pragma unroll
    for (int r = 0; r < 8; ++r) {
        const int mr = r + hi * 8;
        float v = acc[r] * dinv[mr];
        float o = v > 0.0f ? v : (__expf(v) - 1.0f);   // ELU
        out[(size_t)(row0 + mr) * OUT_C + (n0 + cn)] = o;
    }
}

// -------------------------------------------------------------------------
extern "C" void kernel_launch(void* const* d_in, const int* in_sizes, int n_in,
                              void* d_out, int out_size, void* d_ws, size_t ws_size,
                              hipStream_t stream) {
    const float* H   = (const float*)d_in[0];
    const int*   adj = (const int*)d_in[1];
    const float* W   = (const float*)d_in[2];
    const float* a   = (const float*)d_in[3];
    float*       out = (float*)d_out;

    char* ws = (char*)d_ws;
    _Float16* WhT = (_Float16*)ws;                               // 512 KB
    float*    f1  = (float*)(ws + (size_t)OUT_C * N_NODES * 2);  // 16 KB
    float*    f2  = f1 + N_NODES;                                // 16 KB

    gat_k1<<<N_NODES, 64, 0, stream>>>(H, W, a, WhT, f1, f2);
    gat_k2<<<N_NODES / 16, 128, 0, stream>>>(adj, f1, f2, WhT, out);
}
